// RBF_54047868452973
// MI455X (gfx1250) — matrix operand order repaired and checked by reading, not compiled, not run on hardware
//
#include <hip/hip_runtime.h>

// out[b,o] = ||x_b||^2 + ||w_o||^2 - 2 * (x @ w)[b,o]
// Split-f16 WMMA GEMM (hi/lo 3-term emulation, f32 accumulate), register->LDS
// double buffering (1 barrier / k-slab), fused distance epilogue.

typedef __attribute__((ext_vector_type(16))) _Float16 v16h;
typedef __attribute__((ext_vector_type(8)))  float    v8f;

#define M_DIM 16384
#define K_DIM 1024
#define N_DIM 2048

#define BM 128
#define BN 128
#define BK 32
#define SA 40   // LDS row stride in halves for A tiles (32 -> 40 pad vs bank conflicts)
#define SB 40   // LDS row stride in halves for B tiles (stored transposed [n][k])

__device__ __forceinline__ unsigned pack2(_Float16 a, _Float16 b) {
  union { _Float16 h[2]; unsigned u; } p;
  p.h[0] = a; p.h[1] = b;
  return p.u;
}

union V16 { v16h v; uint4 q[2]; };

// ---------------- ||x_b||^2 : one wave32 per row ----------------
__global__ __launch_bounds__(256) void rbf_rowsq(const float* __restrict__ x,
                                                 float* __restrict__ xsq) {
  const int lane = threadIdx.x & 31;
  const int row  = blockIdx.x * 8 + (threadIdx.x >> 5);
  const float* xr = x + (size_t)row * K_DIM;
  float s = 0.0f;
  for (int i = lane * 4; i < K_DIM; i += 128) {
    const float4 v = *(const float4*)(xr + i);
    s += v.x * v.x + v.y * v.y + v.z * v.z + v.w * v.w;
  }
#pragma unroll
  for (int off = 16; off > 0; off >>= 1) s += __shfl_xor(s, off, 32);
  if (lane == 0) xsq[row] = s;
}

// ---------------- ||w_o||^2 : one thread per column ----------------
__global__ __launch_bounds__(256) void rbf_colsq(const float* __restrict__ w,
                                                 float* __restrict__ wsq) {
  const int o = blockIdx.x * 256 + threadIdx.x;
  float s = 0.0f;
  for (int kk = 0; kk < K_DIM; ++kk) {
    const float v = w[(size_t)kk * N_DIM + o];
    s += v * v;
  }
  wsq[o] = s;
}

// ---------------- main fused GEMM + distance epilogue ----------------
__global__ __launch_bounds__(256) void rbf_gemm(const float* __restrict__ x,
                                                const float* __restrict__ w,
                                                const float* __restrict__ xsq,
                                                const float* __restrict__ wsq,
                                                float* __restrict__ out) {
  __shared__ _Float16 Ahi[2][BM * SA];
  __shared__ _Float16 Alo[2][BM * SA];
  __shared__ _Float16 Bhi[2][BN * SB];
  __shared__ _Float16 Blo[2][BN * SB];   // 80 KB total -> 4 WGs / 320 KB WGP

  const int t    = threadIdx.x;     // 0..255, 8 waves
  const int lane = t & 31;
  const int wid  = t >> 5;
  const int m0 = blockIdx.y * BM;
  const int n0 = blockIdx.x * BN;
  const int wm = (wid >> 2) * 64;   // wave row offset inside block tile (2 waves in M)
  const int wn = (wid & 3) * 32;    // wave col offset inside block tile (4 waves in N)

  // staging geometry (constant across k)
  const int arow = t >> 3;          // 0..31 (plus p*32)
  const int ac4  = (t & 7) * 4;
  const int bkk0 = (t >> 7) * 2;    // 0 or 2 (plus p*4)
  const int bn   = t & 127;

  float4 ra[4];                     // in-flight A slab (16 f32 / thread)
  float2 rb[8];                     // in-flight B slab (16 f32 / thread)

  auto load_slab = [&](int k0) {    // issue global loads; results land later
#pragma unroll
    for (int p = 0; p < 4; ++p)
      ra[p] = *(const float4*)(x + (size_t)(m0 + p * 32 + arow) * K_DIM + k0 + ac4);
#pragma unroll
    for (int p = 0; p < 8; ++p) {
      const int kk = p * 4 + bkk0;
      const float* wp = w + (size_t)(k0 + kk) * N_DIM + n0 + bn;
      rb[p] = make_float2(wp[0], wp[N_DIM]);
    }
  };

  auto store_slab = [&](int buf) {  // f32 -> split f16 hi/lo, write LDS
#pragma unroll
    for (int p = 0; p < 4; ++p) {
      const float4 v = ra[p];
      const _Float16 h0 = (_Float16)v.x, h1 = (_Float16)v.y,
                     h2 = (_Float16)v.z, h3 = (_Float16)v.w;
      const _Float16 l0 = (_Float16)(v.x - (float)h0), l1 = (_Float16)(v.y - (float)h1),
                     l2 = (_Float16)(v.z - (float)h2), l3 = (_Float16)(v.w - (float)h3);
      const int off = (p * 32 + arow) * SA + ac4;
      *(uint2*)&Ahi[buf][off] = make_uint2(pack2(h0, h1), pack2(h2, h3));
      *(uint2*)&Alo[buf][off] = make_uint2(pack2(l0, l1), pack2(l2, l3));
    }
#pragma unroll
    for (int p = 0; p < 8; ++p) {
      const int kk = p * 4 + bkk0;
      const float2 v = rb[p];
      const _Float16 h0 = (_Float16)v.x, h1 = (_Float16)v.y;
      const _Float16 l0 = (_Float16)(v.x - (float)h0), l1 = (_Float16)(v.y - (float)h1);
      *(unsigned*)&Bhi[buf][bn * SB + kk] = pack2(h0, h1);
      *(unsigned*)&Blo[buf][bn * SB + kk] = pack2(l0, l1);
    }
  };

  v8f acc[4][2] = {};               // 4 M-frags x 2 N-frags of 16x16 f32

  // Fragment addressing per ISA layout tables (CDNA5 ISA §7.12.2):
  // A 16x32 f16: lanes 0-15 hold K chunks {0..7,16..23}; lanes 16-31 {8..15,24..31}.
  // B 32x16 f16: lane n holds K chunks {0..7,16..23} (lanes 0-15) or {8..15,24..31} (lanes 16-31).
  const int aKoff = (lane < 16) ? 0 : 8;
  const int bKoff = (lane < 16) ? 0 : 8;
  const int fr    = lane & 15;

  // prologue: fill buffer 0
  load_slab(0);
  store_slab(0);
  __syncthreads();

  int buf = 0;
  for (int k0 = 0; k0 < K_DIM; k0 += BK) {
    const bool more = (k0 + BK) < K_DIM;
    if (more) load_slab(k0 + BK);   // global loads fly under the WMMAs below

    // ---- load fragments for this k-slab from LDS buf ----
    V16 ah[4], al[4], bh[2], bl[2];
#pragma unroll
    for (int mi = 0; mi < 4; ++mi) {
      const _Float16* ph = &Ahi[buf][(wm + mi * 16 + fr) * SA + aKoff];
      ah[mi].q[0] = *(const uint4*)(ph);
      ah[mi].q[1] = *(const uint4*)(ph + 16);
      const _Float16* pl = &Alo[buf][(wm + mi * 16 + fr) * SA + aKoff];
      al[mi].q[0] = *(const uint4*)(pl);
      al[mi].q[1] = *(const uint4*)(pl + 16);
    }
#pragma unroll
    for (int ni = 0; ni < 2; ++ni) {
      const _Float16* ph = &Bhi[buf][(wn + ni * 16 + fr) * SB + bKoff];
      bh[ni].q[0] = *(const uint4*)(ph);
      bh[ni].q[1] = *(const uint4*)(ph + 16);
      const _Float16* pl = &Blo[buf][(wn + ni * 16 + fr) * SB + bKoff];
      bl[ni].q[0] = *(const uint4*)(pl);
      bl[ni].q[1] = *(const uint4*)(pl + 16);
    }

    // ---- 3-term split-f16: (hi+lo)(hi'+lo') ~= hi*hi' + hi*lo' + lo*hi' ----
#pragma unroll
    for (int mi = 0; mi < 4; ++mi) {
#pragma unroll
      for (int ni = 0; ni < 2; ++ni) {
        acc[mi][ni] = __builtin_amdgcn_wmma_f32_16x16x32_f16(
            false, ah[mi].v, false, bh[ni].v, (short)0, acc[mi][ni], false, false);
        acc[mi][ni] = __builtin_amdgcn_wmma_f32_16x16x32_f16(
            false, ah[mi].v, false, bl[ni].v, (short)0, acc[mi][ni], false, false);
        acc[mi][ni] = __builtin_amdgcn_wmma_f32_16x16x32_f16(
            false, al[mi].v, false, bh[ni].v, (short)0, acc[mi][ni], false, false);
      }
    }

    if (more) {
      store_slab(buf ^ 1);          // writes hit the *other* buffer: no read conflict
      __syncthreads();              // single barrier per k-slab
    }
    buf ^= 1;
  }

  // ---- epilogue: out = xsq[m] + wsq[n] - 2*cross ; coalesced b32 stores ----
  const int hb = (lane >> 4) << 3;  // upper half-wave holds rows M+8..M+15
#pragma unroll
  for (int mi = 0; mi < 4; ++mi) {
#pragma unroll
    for (int ni = 0; ni < 2; ++ni) {
      const int nn = n0 + wn + ni * 16 + fr;
      const float wsn = wsq[nn];
#pragma unroll
      for (int r = 0; r < 8; ++r) {
        const int mm = m0 + wm + mi * 16 + hb + r;
        out[(size_t)mm * N_DIM + nn] = xsq[mm] + wsn - 2.0f * acc[mi][ni][r];
      }
    }
  }
}

extern "C" void kernel_launch(void* const* d_in, const int* in_sizes, int n_in,
                              void* d_out, int out_size, void* d_ws, size_t ws_size,
                              hipStream_t stream) {
  (void)in_sizes; (void)n_in; (void)out_size; (void)ws_size;
  const float* x = (const float*)d_in[0];   // [16384, 1024] f32
  const float* w = (const float*)d_in[1];   // [1024, 2048] f32
  float* out = (float*)d_out;               // [16384, 2048] f32
  float* xsq = (float*)d_ws;                // [16384] f32
  float* wsq = xsq + M_DIM;                 // [2048]  f32  (ws: ~72 KB used)

  rbf_rowsq<<<M_DIM / 8, 256, 0, stream>>>(x, xsq);
  rbf_colsq<<<N_DIM / 256, 256, 0, stream>>>(w, wsq);

  dim3 grid(N_DIM / BN, M_DIM / BM);        // 16 x 128 = 2048 workgroups
  rbf_gemm<<<grid, 256, 0, stream>>>(x, w, xsq, wsq, out);
}
